// MixtralMoE_44659069944439
// MI455X (gfx1250) — matrix operand order repaired and checked by reading, not compiled, not run on hardware
//
#include <hip/hip_runtime.h>
#include <math.h>

#define T_TOK 2048
#define H_DIM 2048
#define I_DIM 5632
#define E_NUM 8
#define NPAIR (T_TOK * 2)

#define BM 128
#define BN 64
#define BK 32
#define LDK 40          // padded LDS row stride (bf16 elems): 80B -> conflict-free frag reads
#define MAXMT (T_TOK / BM)   // 16 worst-case M tiles per expert

typedef __attribute__((ext_vector_type(16))) __bf16   v16bf;
typedef __attribute__((ext_vector_type(8)))  float    v8f;
typedef __attribute__((ext_vector_type(4)))  unsigned v4u;
typedef __attribute__((ext_vector_type(4)))  float    v4f;

union BFrag { v4u u[2]; v16bf v; };

__device__ __forceinline__ unsigned bf16_bits(float f) {
  unsigned u = __float_as_uint(f);
  return (u + 0x7fffu + ((u >> 16) & 1u)) >> 16;   // round-to-nearest-even
}
__device__ __forceinline__ unsigned pack2(float a, float b) {
  return bf16_bits(a) | (bf16_bits(b) << 16);
}
__device__ __forceinline__ float silu_mul(float g, float u) {
  // silu(g)*u with v_rcp_f32 instead of IEEE division
  float s = __builtin_amdgcn_rcpf(1.f + __expf(-g));
  return g * s * u;
}
// pack 8 fp32 -> 8 bf16 and store 16B to LDS
__device__ __forceinline__ void store_w8(unsigned short* lds, int idx, v4f a0, v4f a1) {
  v4u p;
  p[0] = pack2(a0[0], a0[1]); p[1] = pack2(a0[2], a0[3]);
  p[2] = pack2(a1[0], a1[1]); p[3] = pack2(a1[2], a1[3]);
  *(v4u*)&lds[idx] = p;
}

// A fragment, 16x32 bf16 (MxK). lanes0-15: m=lane, k=[0..7],[16..23]; lanes16-31: m=lane-16, k=[8..15],[24..31]
__device__ __forceinline__ v16bf frag_A(const unsigned short* lds, int m0, int lane) {
  int m  = m0 + (lane & 15);
  int kb = (lane < 16) ? 0 : 8;
  const char* p = (const char*)lds + (size_t)(m * LDK + kb) * 2;
  BFrag f;
  f.u[0] = *(const v4u*)p;          // k = kb .. kb+7
  f.u[1] = *(const v4u*)(p + 32);   // k = kb+16 .. kb+23
  return f.v;
}
// B fragment, 32x16 bf16 (KxN), LDS holds W as [N][K]. lanes0-15: n=lane, k=[0..7],[16..23]; lanes16-31: n=lane-16, k=[8..15],[24..31]
__device__ __forceinline__ v16bf frag_B(const unsigned short* lds, int n0, int lane) {
  int n  = n0 + (lane & 15);
  int kb = (lane < 16) ? 0 : 8;
  const char* p = (const char*)lds + (size_t)(n * LDK + kb) * 2;
  BFrag f;
  f.u[0] = *(const v4u*)p;
  f.u[1] = *(const v4u*)(p + 32);
  return f.v;
}

// ---------------- Router: logits -> softmax -> top2 -> renorm ----------------
__global__ __launch_bounds__(256) void router_kernel(const float* __restrict__ x,
                                                     const float* __restrict__ gw,
                                                     int* __restrict__ tidx,
                                                     float* __restrict__ tw) {
  int t = blockIdx.x, tid = threadIdx.x;
  __shared__ float red[E_NUM][256];
  float acc[E_NUM];
#pragma unroll
  for (int e = 0; e < E_NUM; ++e) acc[e] = 0.f;
  for (int h = tid; h < H_DIM; h += 256) {
    float xv = x[(size_t)t * H_DIM + h];
#pragma unroll
    for (int e = 0; e < E_NUM; ++e) acc[e] += xv * gw[(size_t)e * H_DIM + h];
  }
#pragma unroll
  for (int e = 0; e < E_NUM; ++e) red[e][tid] = acc[e];
  __syncthreads();
  for (int s = 128; s > 0; s >>= 1) {
    if (tid < s)
#pragma unroll
      for (int e = 0; e < E_NUM; ++e) red[e][tid] += red[e][tid + s];
    __syncthreads();
  }
  if (tid == 0) {
    float l[E_NUM], m = -3.4e38f;
#pragma unroll
    for (int e = 0; e < E_NUM; ++e) { l[e] = red[e][0]; m = fmaxf(m, l[e]); }
    float p[E_NUM], s = 0.f;
#pragma unroll
    for (int e = 0; e < E_NUM; ++e) { p[e] = __expf(l[e] - m); s += p[e]; }
#pragma unroll
    for (int e = 0; e < E_NUM; ++e) p[e] /= s;
    int i0 = 0;
    for (int e = 1; e < E_NUM; ++e) if (p[e] > p[i0]) i0 = e;   // strict > keeps lowest idx on ties
    int i1 = (i0 == 0) ? 1 : 0;
    for (int e = 0; e < E_NUM; ++e) if (e != i0 && p[e] > p[i1]) i1 = e;
    float s2 = p[i0] + p[i1];
    tidx[2 * t] = i0; tidx[2 * t + 1] = i1;
    tw[2 * t] = p[i0] / s2; tw[2 * t + 1] = p[i1] / s2;
  }
}

// ------- Compaction: per-expert counts, bases, row assignment (single block) -------
__global__ __launch_bounds__(256) void compact_kernel(const int* __restrict__ tidx,
                                                      int* __restrict__ token_of_row,
                                                      int* __restrict__ row_of_pair,
                                                      int* __restrict__ cntArr,
                                                      int* __restrict__ baseArr) {
  __shared__ int scnt[E_NUM], sbase[E_NUM], scur[E_NUM];
  int tid = threadIdx.x;
  if (tid < E_NUM) scnt[tid] = 0;
  __syncthreads();
  for (int p = tid; p < NPAIR; p += 256) atomicAdd(&scnt[tidx[p]], 1);
  __syncthreads();
  if (tid == 0) {
    int b = 0;
    for (int e = 0; e < E_NUM; ++e) { sbase[e] = b; b += scnt[e]; }
  }
  __syncthreads();
  if (tid < E_NUM) { scur[tid] = sbase[tid]; cntArr[tid] = scnt[tid]; baseArr[tid] = sbase[tid]; }
  __syncthreads();
  for (int p = tid; p < NPAIR; p += 256) {
    int e = tidx[p];
    int r = atomicAdd(&scur[e], 1);     // row order inside expert is irrelevant to final output
    token_of_row[r] = p >> 1;
    row_of_pair[p]  = r;
  }
}

// ---------------- x fp32 -> bf16 ----------------
__global__ __launch_bounds__(256) void cvt_x_kernel(const float* __restrict__ x,
                                                    unsigned* __restrict__ xb) {
  size_t i = (size_t)blockIdx.x * blockDim.x + threadIdx.x;   // 4 floats per thread
  v4f v = *(const v4f*)(x + i * 4);
  xb[i * 2]     = pack2(v[0], v[1]);
  xb[i * 2 + 1] = pack2(v[2], v[3]);
}

// ------- G1: A[row,i] = silu(x·w1) * (x·w3), bf16 out, WMMA bf16, double-buffered LDS -------
__global__ __launch_bounds__(256) void gateup_kernel(const unsigned short* __restrict__ xb,
                                                     const float* __restrict__ w1,
                                                     const float* __restrict__ w3,
                                                     const int* __restrict__ token_of_row,
                                                     const int* __restrict__ cntArr,
                                                     const int* __restrict__ baseArr,
                                                     unsigned short* __restrict__ Abf) {
  int e = blockIdx.x / MAXMT, mt = blockIdx.x % MAXMT;
  int cntE = cntArr[e];
  if (mt * BM >= cntE) return;
  int base = baseArr[e];
  int n0 = blockIdx.y * BN;
  int tid = threadIdx.x, lane = tid & 31, wave = tid >> 5;
  int wm = wave >> 1, wn = wave & 1;             // 4 waves along M, 2 along N

  __shared__ unsigned short lds_x[2][BM * LDK];
  __shared__ unsigned short lds_w1[2][BN * LDK];
  __shared__ unsigned short lds_w3[2][BN * LDK];

  // x loader: 2 threads/row, 8 bf16 (16B) each
  int lrow = tid >> 1, lhalf = tid & 1;
  int gm = mt * BM + lrow;
  int tok = (gm < cntE) ? token_of_row[base + gm] : 0;
  const unsigned short* xsrc = xb + (size_t)tok * H_DIM + lhalf * 16;
  int xIdx = lrow * LDK + lhalf * 16;
  // weight loader: 4 threads/row, 8 floats each
  int wrow = tid >> 2, wq = tid & 3;
  const float* w1src = w1 + ((size_t)e * I_DIM + (n0 + wrow)) * H_DIM + wq * 8;
  const float* w3src = w3 + ((size_t)e * I_DIM + (n0 + wrow)) * H_DIM + wq * 8;
  int wIdx = wrow * LDK + wq * 8;

  v8f accg[2][2] = {}; v8f accu[2][2] = {};

  // prologue: chunk 0 -> buffer 0
  {
    v4u xv = *(const v4u*)(xsrc);
    v4f a0 = *(const v4f*)(w1src), a1 = *(const v4f*)(w1src + 4);
    v4f b0 = *(const v4f*)(w3src), b1 = *(const v4f*)(w3src + 4);
    *(v4u*)&lds_x[0][xIdx] = xv;
    store_w8(lds_w1[0], wIdx, a0, a1);
    store_w8(lds_w3[0], wIdx, b0, b1);
  }
  __syncthreads();

  int buf = 0;
  for (int k0 = 0; k0 < H_DIM; k0 += BK) {
    int kn = k0 + BK;
    // issue next chunk's global loads first (latency hides under WMMA)
    v4u xv = {}; v4f a0 = {}, a1 = {}, b0 = {}, b1 = {};
    if (kn < H_DIM) {
      xv = *(const v4u*)(xsrc + kn);
      a0 = *(const v4f*)(w1src + kn); a1 = *(const v4f*)(w1src + kn + 4);
      b0 = *(const v4f*)(w3src + kn); b1 = *(const v4f*)(w3src + kn + 4);
    }
    // compute on current buffer
    v16bf b1f[2], b3f[2];
#pragma unroll
    for (int fn = 0; fn < 2; ++fn) {
      b1f[fn] = frag_B(lds_w1[buf], wn * 32 + fn * 16, lane);
      b3f[fn] = frag_B(lds_w3[buf], wn * 32 + fn * 16, lane);
    }
#pragma unroll
    for (int fm = 0; fm < 2; ++fm) {
      v16bf af = frag_A(lds_x[buf], wm * 32 + fm * 16, lane);
#pragma unroll
      for (int fn = 0; fn < 2; ++fn) {
        accg[fm][fn] = __builtin_amdgcn_wmma_f32_16x16x32_bf16(
            false, af, false, b1f[fn], (short)0, accg[fm][fn], false, false);
        accu[fm][fn] = __builtin_amdgcn_wmma_f32_16x16x32_bf16(
            false, af, false, b3f[fn], (short)0, accu[fm][fn], false, false);
      }
    }
    // stage next chunk into the other buffer; one barrier per K-chunk
    if (kn < H_DIM) {
      *(v4u*)&lds_x[buf ^ 1][xIdx] = xv;
      store_w8(lds_w1[buf ^ 1], wIdx, a0, a1);
      store_w8(lds_w3[buf ^ 1], wIdx, b0, b1);
      __syncthreads();
    }
    buf ^= 1;
  }

  // epilogue: silu(g)*u -> bf16. C layout: vgpr r -> m = r + (lane<16?0:8), n = lane&15
  int mHi = (lane < 16) ? 0 : 8;
  int nLane = lane & 15;
  bool full = (mt * BM + BM) <= cntE;   // uniform: interior tiles take branch-free path
#pragma unroll
  for (int fm = 0; fm < 2; ++fm) {
#pragma unroll
    for (int fn = 0; fn < 2; ++fn) {
      int mB = mt * BM + wm * 32 + fm * 16 + mHi;
      int col = n0 + wn * 32 + fn * 16 + nLane;
      unsigned short* dst = Abf + (size_t)(base + mB) * I_DIM + col;
      if (full) {
#pragma unroll
        for (int r = 0; r < 8; ++r)
          dst[(size_t)r * I_DIM] =
              (unsigned short)bf16_bits(silu_mul(accg[fm][fn][r], accu[fm][fn][r]));
      } else {
#pragma unroll
        for (int r = 0; r < 8; ++r)
          if (mB + r < cntE)
            dst[(size_t)r * I_DIM] =
                (unsigned short)bf16_bits(silu_mul(accg[fm][fn][r], accu[fm][fn][r]));
      }
    }
  }
}

// ------- G2: Y[row,h] = A[row,:] · w2[e,h,:], fp32 out, WMMA bf16, double-buffered LDS -------
__global__ __launch_bounds__(256) void down_kernel(const unsigned short* __restrict__ Abf,
                                                   const float* __restrict__ w2,
                                                   const int* __restrict__ cntArr,
                                                   const int* __restrict__ baseArr,
                                                   float* __restrict__ Y) {
  int e = blockIdx.x / MAXMT, mt = blockIdx.x % MAXMT;
  int cntE = cntArr[e];
  if (mt * BM >= cntE) return;
  int base = baseArr[e];
  int n0 = blockIdx.y * BN;
  int tid = threadIdx.x, lane = tid & 31, wave = tid >> 5;
  int wm = wave >> 1, wn = wave & 1;

  __shared__ unsigned short lds_a[2][BM * LDK];
  __shared__ unsigned short lds_w[2][BN * LDK];

  int lrow = tid >> 1, lhalf = tid & 1;
  int aRow = base + mt * BM + lrow;
  if (aRow > NPAIR - 1) aRow = NPAIR - 1;        // clamp OOB tail rows (results discarded)
  const unsigned short* asrc = Abf + (size_t)aRow * I_DIM + lhalf * 16;
  int aIdx = lrow * LDK + lhalf * 16;
  int wrow = tid >> 2, wq = tid & 3;
  const float* wsrc = w2 + ((size_t)e * H_DIM + (n0 + wrow)) * I_DIM + wq * 8;
  int wIdx = wrow * LDK + wq * 8;

  v8f acc[2][2] = {};

  {
    v4u av = *(const v4u*)(asrc);
    v4f a0 = *(const v4f*)(wsrc), a1 = *(const v4f*)(wsrc + 4);
    *(v4u*)&lds_a[0][aIdx] = av;
    store_w8(lds_w[0], wIdx, a0, a1);
  }
  __syncthreads();

  int buf = 0;
  for (int k0 = 0; k0 < I_DIM; k0 += BK) {
    int kn = k0 + BK;
    v4u av = {}; v4f a0 = {}, a1 = {};
    if (kn < I_DIM) {
      av = *(const v4u*)(asrc + kn);
      a0 = *(const v4f*)(wsrc + kn); a1 = *(const v4f*)(wsrc + kn + 4);
    }
    v16bf bf[2];
#pragma unroll
    for (int fn = 0; fn < 2; ++fn) bf[fn] = frag_B(lds_w[buf], wn * 32 + fn * 16, lane);
#pragma unroll
    for (int fm = 0; fm < 2; ++fm) {
      v16bf af = frag_A(lds_a[buf], wm * 32 + fm * 16, lane);
#pragma unroll
      for (int fn = 0; fn < 2; ++fn)
        acc[fm][fn] = __builtin_amdgcn_wmma_f32_16x16x32_bf16(
            false, af, false, bf[fn], (short)0, acc[fm][fn], false, false);
    }
    if (kn < I_DIM) {
      *(v4u*)&lds_a[buf ^ 1][aIdx] = av;
      store_w8(lds_w[buf ^ 1], wIdx, a0, a1);
      __syncthreads();
    }
    buf ^= 1;
  }

  int mHi = (lane < 16) ? 0 : 8;
  int nLane = lane & 15;
  bool full = (mt * BM + BM) <= cntE;
#pragma unroll
  for (int fm = 0; fm < 2; ++fm) {
#pragma unroll
    for (int fn = 0; fn < 2; ++fn) {
      int mB = mt * BM + wm * 32 + fm * 16 + mHi;
      int col = n0 + wn * 32 + fn * 16 + nLane;
      float* dst = Y + (size_t)(base + mB) * H_DIM + col;
      if (full) {
#pragma unroll
        for (int r = 0; r < 8; ++r) dst[(size_t)r * H_DIM] = acc[fm][fn][r];
      } else {
#pragma unroll
        for (int r = 0; r < 8; ++r)
          if (mB + r < cntE) dst[(size_t)r * H_DIM] = acc[fm][fn][r];
      }
    }
  }
}

// ---------------- Weighted combine of the two expert rows per token ----------------
__global__ __launch_bounds__(256) void combine_kernel(const float* __restrict__ Y,
                                                      const int* __restrict__ row_of_pair,
                                                      const float* __restrict__ tw,
                                                      float* __restrict__ out) {
  int t = blockIdx.x;
  int r0 = row_of_pair[2 * t], r1 = row_of_pair[2 * t + 1];
  float w0 = tw[2 * t], w1v = tw[2 * t + 1];
  const v4f* y0 = (const v4f*)(Y + (size_t)r0 * H_DIM);
  const v4f* y1 = (const v4f*)(Y + (size_t)r1 * H_DIM);
  v4f* o = (v4f*)(out + (size_t)t * H_DIM);
  for (int h = threadIdx.x; h < H_DIM / 4; h += 256)
    o[h] = w0 * y0[h] + w1v * y1[h];
}

extern "C" void kernel_launch(void* const* d_in, const int* in_sizes, int n_in,
                              void* d_out, int out_size, void* d_ws, size_t ws_size,
                              hipStream_t stream) {
  (void)in_sizes; (void)n_in; (void)out_size; (void)ws_size;
  const float* x  = (const float*)d_in[0];
  const float* gw = (const float*)d_in[1];
  const float* w1 = (const float*)d_in[2];
  const float* w3 = (const float*)d_in[3];
  const float* w2 = (const float*)d_in[4];
  float* out = (float*)d_out;

  char* ws = (char*)d_ws;
  int*            topk_idx     = (int*)(ws + 0);           // 4096 ints
  float*          topk_w       = (float*)(ws + 16384);     // 4096 floats
  int*            token_of_row = (int*)(ws + 32768);       // 4096 ints
  int*            row_of_pair  = (int*)(ws + 49152);       // 4096 ints
  int*            cntArr       = (int*)(ws + 65536);       // 8 ints
  int*            baseArr      = (int*)(ws + 65568);       // 8 ints
  unsigned short* xb           = (unsigned short*)(ws + 65600);     // 2048*2048 bf16 = 8 MB
  unsigned short* Abf          = (unsigned short*)(ws + 8454208);   // 4096*5632 bf16 = 46.1 MB
  float*          Y            = (float*)(ws + 54591552);           // 4096*2048 f32  = 33.6 MB

  router_kernel<<<T_TOK, 256, 0, stream>>>(x, gw, topk_idx, topk_w);
  compact_kernel<<<1, 256, 0, stream>>>(topk_idx, token_of_row, row_of_pair, cntArr, baseArr);
  cvt_x_kernel<<<(T_TOK * H_DIM) / (256 * 4), 256, 0, stream>>>(x, (unsigned*)xb);
  gateup_kernel<<<dim3(E_NUM * MAXMT, I_DIM / BN), 256, 0, stream>>>(
      xb, w1, w3, token_of_row, cntArr, baseArr, Abf);
  down_kernel<<<dim3(E_NUM * MAXMT, H_DIM / BN), 256, 0, stream>>>(
      Abf, w2, cntArr, baseArr, Y);
  combine_kernel<<<T_TOK, 256, 0, stream>>>(Y, row_of_pair, topk_w, out);
}
